// NeuralObliviousDecisionTree_34428457845294
// MI455X (gfx1250) — matrix operand order repaired and checked by reading, not compiled, not run on hardware
//
#include <hip/hip_runtime.h>

#define N_FEATURES 512
#define DEPTH 6
#define N_TREES 64
#define D_OUT 64
#define N_LEAVES 64
#define BATCH_TILE 16

typedef __attribute__((ext_vector_type(16))) _Float16 v16h;
typedef __attribute__((ext_vector_type(8)))  float    v8f;

union Frag { v16h v; uint4 q[2]; };

// Load a 16xf16 WMMA fragment as two contiguous 16-byte chunks.
__device__ __forceinline__ v16h load_frag(const _Float16* p0, const _Float16* p1) {
    Frag f;
    f.q[0] = *(const uint4*)p0;
    f.q[1] = *(const uint4*)p1;
    return f.v;
}

// ---- prep: W (6,512,64) f32 -> Wt[j=d*64+t][f] f16 (column-major for B frags) ----
__global__ __launch_bounds__(256)
void prep_w_kernel(const float* __restrict__ W, _Float16* __restrict__ Wt) {
    const int i = blockIdx.x * 256 + threadIdx.x;   // 384*512 = 196608 threads
    const int j = i >> 9;                           // column 0..383 (d*64+t)
    const int f = i & 511;
    Wt[i] = (_Float16)W[(j >> 6) * (N_FEATURES * N_TREES) + f * N_TREES + (j & 63)];
}

// ---- prep: leaf (64,64,64) t,l,d f32 -> leafT[t][d][l] f16 (l contiguous = K-major) ----
__global__ __launch_bounds__(256)
void prep_leaf_kernel(const float* __restrict__ leaf, _Float16* __restrict__ leafT) {
    const int i = blockIdx.x * 256 + threadIdx.x;   // 64*64*64 = 262144 threads
    const int t = i >> 12;
    const int n = (i >> 6) & 63;                    // output dim
    const int l = i & 63;                           // leaf (K)
    leafT[i] = (_Float16)leaf[t * (N_LEAVES * D_OUT) + l * D_OUT + n];
}

// Fused NODT forward: split = sigmoid(x@W+b); path = level products; out = path@leaf / 64
__global__ __launch_bounds__(256)
void nodt_fused_kernel(const float* __restrict__ x,
                       const _Float16* __restrict__ Wt,
                       const float* __restrict__ bias,
                       const _Float16* __restrict__ leafT,
                       float* __restrict__ out)
{
    __shared__ _Float16 xsh[BATCH_TILE * N_FEATURES];          // 16KB f16 x-tile
    __shared__ float    split[BATCH_TILE * DEPTH * N_TREES];   // 24KB
    __shared__ _Float16 pathh[2][BATCH_TILE * N_LEAVES];       // 2x2KB ping-pong
    __shared__ float    red[4 * 256];                          // 4KB K-reduction

    const int tid  = threadIdx.x;
    const int lane = tid & 31;
    const int wid  = tid >> 5;
    const int half = lane >> 4;
    const int ncol = lane & 15;      // N col of 16x16 tile; also M row for A frags
    const int btile = blockIdx.x;
    const long long xbase = (long long)btile * BATCH_TILE * N_FEATURES;

    // ---- stage x-tile as f16 in LDS: coalesced b64 loads, packed f16 b32 stores ----
    {
        const float2* x2 = (const float2*)(x + xbase);
        unsigned int* xsu = (unsigned int*)xsh;
        #pragma unroll
        for (int i = 0; i < (BATCH_TILE * N_FEATURES / 2) / 256; ++i) {
            const float2 v = x2[tid + i * 256];
            union { _Float16 h[2]; unsigned int u; } pk;
            pk.h[0] = (_Float16)v.x;
            pk.h[1] = (_Float16)v.y;
            xsu[tid + i * 256] = pk.u;
        }
    }
    __syncthreads();

    // ================= Step 1: logits = x @ W  (M=16, N=384, K=512) =================
    // 24 column tiles of 16; 8 waves x 3 tiles. A frag shared across a wave's tiles.
    v8f acc[3] = {};
    for (int kb = 0; kb < N_FEATURES / 32; ++kb) {
        // A: 16x32 f16; per lane two contiguous 8-elem K-runs
        const _Float16* ar = xsh + ncol * N_FEATURES + kb * 32 + 8 * half;
        const v16h afrag = load_frag(ar, ar + 16);
        #pragma unroll
        for (int j = 0; j < 3; ++j) {
            const int jb   = wid + j * 8;
            const int jcol = jb * 16 + ncol;
            // B: 32x16 f16; per lane two 8-elem K-runs (8*half, 16+8*half) from column-major Wt
            const _Float16* br = Wt + (long long)jcol * N_FEATURES + kb * 32 + half * 8;
            const v16h bfrag = load_frag(br, br + 16);
            acc[j] = __builtin_amdgcn_wmma_f32_16x16x32_f16(
                false, afrag, false, bfrag, (short)0, acc[j], false, false);
        }
    }

    // bias + sigmoid -> split_probs LDS: split[m][d*64+t]
    #pragma unroll
    for (int j = 0; j < 3; ++j) {
        const int jcol = (wid + j * 8) * 16 + ncol;
        const float bb = bias[jcol];
        #pragma unroll
        for (int r = 0; r < 8; ++r) {
            const int m   = r + 8 * half;
            const float z = acc[j][r] + bb;
            split[m * (DEPTH * N_TREES) + jcol] = 1.0f / (1.0f + __expf(-z));
        }
    }
    __syncthreads();

    // ================= Steps 2+3: out = (path @ leaf) / 64 =================
    const int nt = wid & 3;      // N tile (output dims)
    const int kh = wid >> 2;     // K half of the 64-leaf dimension
    v8f cacc = {};

    for (int t = 0; t < N_TREES; ++t) {
        _Float16* pbuf = pathh[t & 1];
        // path[b][l] = prod_d (bit ? 1-p : p); 1024 elems / 256 threads
        #pragma unroll
        for (int i = 0; i < 4; ++i) {
            const int e  = tid + i * 256;
            const int bl = e >> 6;
            const int l  = e & 63;
            float prod = 1.0f;
            #pragma unroll
            for (int d = 0; d < DEPTH; ++d) {
                const float p = split[bl * (DEPTH * N_TREES) + d * N_TREES + t];
                const int bit = (l >> (DEPTH - 1 - d)) & 1;
                prod *= bit ? (1.0f - p) : p;
            }
            pbuf[e] = (_Float16)prod;
        }
        __syncthreads();

        // A frag from path LDS (two 8-elem runs in this wave's 32-wide K half)
        const _Float16* ar = pbuf + ncol * N_LEAVES + kh * 32 + 8 * half;
        const v16h afrag = load_frag(ar, ar + 16);
        // B frag from leafT (two 8-elem K-runs per lane's output column, matching A)
        const _Float16* br = leafT + (long long)t * (N_LEAVES * D_OUT)
                           + (nt * 16 + ncol) * N_LEAVES + kh * 32 + half * 8;
        if (t + 1 < N_TREES)
            __builtin_prefetch(br + N_LEAVES * D_OUT, 0, 1);
        const v16h bfrag = load_frag(br, br + 16);
        cacc = __builtin_amdgcn_wmma_f32_16x16x32_f16(
            false, afrag, false, bfrag, (short)0, cacc, false, false);
        // ping-pong buffer: the barrier above orders reads of buf[t-1] vs writes in t+1
    }

    // cross-wave K reduction, scale 1/64, store
    if (kh == 1) {
        #pragma unroll
        for (int r = 0; r < 8; ++r)
            red[nt * 256 + r * 32 + lane] = cacc[r];
    }
    __syncthreads();
    if (kh == 0) {
        const float inv = 1.0f / (float)N_TREES;
        #pragma unroll
        for (int r = 0; r < 8; ++r) {
            const float val = (cacc[r] + red[nt * 256 + r * 32 + lane]) * inv;
            const int m = r + 8 * half;
            out[((long long)btile * BATCH_TILE + m) * D_OUT + nt * 16 + ncol] = val;
        }
    }
}

extern "C" void kernel_launch(void* const* d_in, const int* in_sizes, int n_in,
                              void* d_out, int out_size, void* d_ws, size_t ws_size,
                              hipStream_t stream) {
    const float* x    = (const float*)d_in[0];   // (B, 512)
    const float* W    = (const float*)d_in[1];   // (6, 512, 64)
    const float* b    = (const float*)d_in[2];   // (6, 64)
    const float* leaf = (const float*)d_in[3];   // (64, 64, 64)
    float* out        = (float*)d_out;           // (B, 64)

    _Float16* Wt    = (_Float16*)d_ws;                                   // 384*512 f16 = 384KB
    _Float16* leafT = (_Float16*)((char*)d_ws + 384 * 512 * sizeof(_Float16)); // 512KB

    const int B = in_sizes[0] / N_FEATURES;      // 16384
    const int nblocks = B / BATCH_TILE;          // 1024 blocks of 8 wave32s

    prep_w_kernel<<<dim3((384 * 512) / 256), dim3(256), 0, stream>>>(W, Wt);
    prep_leaf_kernel<<<dim3((64 * 64 * 64) / 256), dim3(256), 0, stream>>>(leaf, leafT);
    nodt_fused_kernel<<<dim3(nblocks), dim3(256), 0, stream>>>(x, Wt, b, leafT, out);
}
